// GraphAttentionMLP_33268816675089
// MI455X (gfx1250) — matrix operand order repaired and checked by reading, not compiled, not run on hardware
//
#include <hip/hip_runtime.h>
#include <hip/hip_bf16.h>
#include <stdint.h>

typedef __attribute__((ext_vector_type(16))) __bf16 bf16x16;
typedef __attribute__((ext_vector_type(8)))  float  f32x8;
typedef __attribute__((ext_vector_type(2)))  float  f32x2;
typedef __attribute__((ext_vector_type(2)))  __bf16 bf16x2;

union FragU { uint32_t u[8]; bf16x16 v; };

// native f32 -> bf16 convert (let clang emit v_cvt_*_bf16)
__device__ __forceinline__ uint16_t f2bf(float x){
  union { __bf16 b; uint16_t u; } t;
  t.b = (__bf16)x;
  return t.u;
}
__device__ __forceinline__ float bf2f(uint16_t h){
  return __uint_as_float(((uint32_t)h) << 16);
}
// packed pair convert (v_cvt_pk_bf16_f32 if available)
__device__ __forceinline__ uint32_t pk2(float a, float b){
  f32x2 f; f.x = a; f.y = b;
  bf16x2 p = __builtin_convertvector(f, bf16x2);
  union { bf16x2 v; uint32_t u; } t;
  t.v = p;
  return t.u;
}
__device__ __forceinline__ float silu_f(float x){ return x / (1.0f + __expf(-x)); }
__device__ __forceinline__ float slrelu(float x){
  float s = 1.0f / (1.0f + __expf(-x));
  return 0.6f * x + 0.4f * x * (2.0f * s - 1.0f);
}
// monotone float<->uint mapping for atomicMax on floats
__device__ __forceinline__ uint32_t f2mono(float f){
  uint32_t u = __float_as_uint(f);
  return (u & 0x80000000u) ? ~u : (u | 0x80000000u);
}
__device__ __forceinline__ float mono2f(uint32_t u){
  return __uint_as_float((u & 0x80000000u) ? (u & 0x7FFFFFFFu) : ~u);
}
__device__ __forceinline__ void lds_fence(){
  asm volatile("s_wait_dscnt 0" ::: "memory");
}

// B-fragment: packed layout, one 256-dword block per (32K x 16N) tile.
// lane l holds n = n0 + (l&15); VGPR j holds K pair k = k0 + (l>>4)*16 + 2j.
__device__ __forceinline__ bf16x16 load_bfrag(const uint32_t* __restrict__ pkd, int blk, int lane){
  FragU f;
  const uint32_t* p = pkd + (size_t)blk * 256 + lane * 8;
#pragma unroll
  for (int i = 0; i < 8; ++i) f.u[i] = p[i];
  return f.v;
}

// A-fragment gather from LDS bf16 tile (viewed as dwords), contiguous k order (same as load_bfrag):
// lane<16: VGPR0-7 K=0..15; lane>=16: K=16..31.
__device__ __forceinline__ bf16x16 gather_afrag(const uint32_t* __restrict__ lds, int pitch_dw, int k0, int lane){
  int m = lane & 15, hl = lane >> 4;
  FragU f;
#pragma unroll
  for (int j = 0; j < 8; ++j){
    int kl = 2*j + hl * 16;
    f.u[j] = lds[m * pitch_dw + ((k0 + kl) >> 1)];
  }
  return f.v;
}

__device__ __forceinline__ f32x8 wmma_bf16(bf16x16 a, bf16x16 b, f32x8 c){
  return __builtin_amdgcn_wmma_f32_16x16x32_bf16(false, a, false, b, (short)0, c, false, false);
}

// ---------------- weight packing: f32 [K,N] row-major -> bf16 fragment blocks -------------
__global__ void pack_weight_kernel(const float* __restrict__ W, uint32_t* __restrict__ dst,
                                   int Kpad, int N, int Kreal){
  int total = (Kpad >> 5) * (N >> 4) * 256;
  int t = blockIdx.x * 256 + threadIdx.x;
  if (t >= total) return;
  int b = t >> 8, r = t & 255;
  int l = r >> 3, j = r & 7;
  int nnb = N >> 4;
  int kb = b / nnb, nb = b - kb * nnb;
  int n = nb * 16 + (l & 15);
  int k = kb * 32 + ((l >> 4) * 16) + 2 * j;
  float a0 = (k     < Kreal) ? W[(size_t)k * N + n]       : 0.0f;
  float a1 = (k + 1 < Kreal) ? W[(size_t)(k + 1) * N + n] : 0.0f;
  dst[t] = pk2(a0, a1);
}

__global__ void zero_kernel(uint32_t* __restrict__ p, int n){
  int t = blockIdx.x * 256 + threadIdx.x;
  if (t < n) p[t] = 0u;
}

// ---------------- per-edge fused kernel: one wave per 16-edge tile ------------------------
#define WV_BYTES 29696
__global__ void edge_kernel(
    const float* __restrict__ message, const float* __restrict__ edge_attr,
    const float* __restrict__ edge_scalars,
    const float* __restrict__ b1, const float* __restrict__ b2,
    const float* __restrict__ alpha_dot, const int* __restrict__ edge_dst,
    const uint32_t* __restrict__ w1p, const uint32_t* __restrict__ w2p, const uint32_t* __restrict__ w3p,
    const uint32_t* __restrict__ eap, const uint32_t* __restrict__ ea2p,
    const uint32_t* __restrict__ alphap, const uint32_t* __restrict__ linp, const uint32_t* __restrict__ valp,
    uint16_t* __restrict__ v_ws, float* __restrict__ la_ws, uint32_t* __restrict__ m_mono,
    int n_tiles){
  extern __shared__ char smem[];
  int lane = threadIdx.x & 31;
  int wv   = threadIdx.x >> 5;
  int tile = blockIdx.x * 2 + wv;
  if (tile >= n_tiles) return;
  int e0 = tile * 16;
  char* wb = smem + (size_t)wv * WV_BYTES;
  uint16_t* sbuf  = (uint16_t*)(wb);          // 16x64 bf16
  uint16_t* hbuf  = (uint16_t*)(wb + 2048);   // 16x64 bf16
  uint16_t* attrb = (uint16_t*)(wb + 4096);   // 16x32 bf16 (zero padded K)
  uint16_t* eab   = (uint16_t*)(wb + 5120);   // 16x256 bf16 (later reused as f32 la scratch)
  uint16_t* ea2b  = (uint16_t*)(wb + 13312);  // 16x256 bf16 (later overwritten with gated)
  uint16_t* msgb  = (uint16_t*)(wb + 21504);  // 16x256 bf16
  int colb = lane & 15, hl = lane >> 4;

  // stage edge_scalars tile as bf16
  for (int i = lane; i < 16*64; i += 32){
    int mr = i >> 6, c = i & 63;
    sbuf[i] = f2bf(edge_scalars[(size_t)(e0 + mr) * 64 + c]);
  }
  lds_fence();

  // h1 = silu(s @ w1 + b1)   [16,64]
  {
    bf16x16 a0 = gather_afrag((const uint32_t*)sbuf, 32, 0,  lane);
    bf16x16 a1 = gather_afrag((const uint32_t*)sbuf, 32, 32, lane);
#pragma unroll
    for (int nt = 0; nt < 4; ++nt){
      f32x8 c = {};
      c = wmma_bf16(a0, load_bfrag(w1p, 0*4 + nt, lane), c);
      c = wmma_bf16(a1, load_bfrag(w1p, 1*4 + nt, lane), c);
      float bias = b1[nt*16 + colb];
#pragma unroll
      for (int r = 0; r < 8; ++r)
        hbuf[(r + hl*8)*64 + nt*16 + colb] = f2bf(silu_f(c[r] + bias));
    }
  }
  lds_fence();

  // h2 = silu(h1 @ w2 + b2)  [16,64]  (write back into sbuf)
  {
    bf16x16 a0 = gather_afrag((const uint32_t*)hbuf, 32, 0,  lane);
    bf16x16 a1 = gather_afrag((const uint32_t*)hbuf, 32, 32, lane);
#pragma unroll
    for (int nt = 0; nt < 4; ++nt){
      f32x8 c = {};
      c = wmma_bf16(a0, load_bfrag(w2p, 0*4 + nt, lane), c);
      c = wmma_bf16(a1, load_bfrag(w2p, 1*4 + nt, lane), c);
      float bias = b2[nt*16 + colb];
#pragma unroll
      for (int r = 0; r < 8; ++r)
        sbuf[(r + hl*8)*64 + nt*16 + colb] = f2bf(silu_f(c[r] + bias));
    }
  }

  // stage edge_attr tile [16,32] bf16, K 16..31 zero
  for (int i = lane; i < 16*32; i += 32){
    int mr = i >> 5, c = i & 31;
    attrb[i] = (c < 16) ? f2bf(edge_attr[(size_t)(e0 + mr) * 16 + c]) : (uint16_t)0;
  }
  lds_fence();

  // ea = attr @ W_ea ; ea2 = attr @ W_ea2   [16,256] each (zero-padded K=32 WMMA)
  {
    bf16x16 aa = gather_afrag((const uint32_t*)attrb, 16, 0, lane);
#pragma unroll
    for (int nt = 0; nt < 16; ++nt){
      f32x8 c1 = {}; c1 = wmma_bf16(aa, load_bfrag(eap,  nt, lane), c1);
      f32x8 c2 = {}; c2 = wmma_bf16(aa, load_bfrag(ea2p, nt, lane), c2);
      int n = nt*16 + colb;
#pragma unroll
      for (int r = 0; r < 8; ++r){
        int mr = r + hl*8;
        eab [mr*256 + n] = f2bf(c1[r]);
        ea2b[mr*256 + n] = f2bf(c2[r]);
      }
    }
  }
  lds_fence();

  // w = h2 @ w3 ; msg = message * w * ea -> bf16 LDS
  {
    bf16x16 a0 = gather_afrag((const uint32_t*)sbuf, 32, 0,  lane);
    bf16x16 a1 = gather_afrag((const uint32_t*)sbuf, 32, 32, lane);
#pragma unroll
    for (int nt = 0; nt < 16; ++nt){
      f32x8 c = {};
      c = wmma_bf16(a0, load_bfrag(w3p, 0*16 + nt, lane), c);
      c = wmma_bf16(a1, load_bfrag(w3p, 1*16 + nt, lane), c);
      int n = nt*16 + colb;
#pragma unroll
      for (int r = 0; r < 8; ++r){
        int mr = r + hl*8;
        float mv = message[(size_t)(e0 + mr) * 256 + n];
        msgb[mr*256 + n] = f2bf(c[r] * mv * bf2f(eab[mr*256 + n]));
      }
    }
  }
  lds_fence();

  // keep all 8 msg A-fragments in registers across the W_alpha and W_lin GEMMs
  bf16x16 am[8];
#pragma unroll
  for (int ks = 0; ks < 8; ++ks) am[ks] = gather_afrag((const uint32_t*)msgb, 128, ks*32, lane);

  // la = reduce_k( slrelu(msg @ W_alpha) * alpha_dot )  [16,8] ; atomic segment max
  {
    float* laS = (float*)eab;   // 16x64 f32 scratch (4KB) — ea no longer needed
#pragma unroll
    for (int nt = 0; nt < 4; ++nt){
      f32x8 c = {};
#pragma unroll
      for (int ks = 0; ks < 8; ++ks) c = wmma_bf16(am[ks], load_bfrag(alphap, ks*4 + nt, lane), c);
      int col = nt*16 + colb;
      float sc = alpha_dot[col];              // flat [h*8+k] == col
#pragma unroll
      for (int r = 0; r < 8; ++r)
        laS[(r + hl*8)*64 + col] = slrelu(c[r]) * sc;
    }
    lds_fence();
#pragma unroll
    for (int t = 0; t < 4; ++t){
      int idx = t*32 + lane;
      int mr = idx >> 3, h = idx & 7;
      float s = 0.f;
#pragma unroll
      for (int kk = 0; kk < 8; ++kk) s += laS[mr*64 + h*8 + kk];
      la_ws[(size_t)(e0 + mr) * 8 + h] = s;
      int dst = edge_dst[e0 + mr];
      atomicMax(&m_mono[(size_t)dst * 8 + h], f2mono(s));
    }
  }

  // gated = silu(msg @ W_lin) * ea2   (overwrite ea2b in place)
#pragma unroll
  for (int nt = 0; nt < 16; ++nt){
    f32x8 c = {};
#pragma unroll
    for (int ks = 0; ks < 8; ++ks) c = wmma_bf16(am[ks], load_bfrag(linp, ks*16 + nt, lane), c);
    int n = nt*16 + colb;
#pragma unroll
    for (int r = 0; r < 8; ++r){
      int mr = r + hl*8;
      float g = silu_f(c[r]) * bf2f(ea2b[mr*256 + n]);
      ea2b[mr*256 + n] = f2bf(g);
    }
  }
  lds_fence();

  // v = gated @ W_val  -> bf16 to global workspace
  {
    bf16x16 ag[8];
#pragma unroll
    for (int ks = 0; ks < 8; ++ks) ag[ks] = gather_afrag((const uint32_t*)ea2b, 128, ks*32, lane);
#pragma unroll
    for (int nt = 0; nt < 16; ++nt){
      f32x8 c = {};
#pragma unroll
      for (int ks = 0; ks < 8; ++ks) c = wmma_bf16(ag[ks], load_bfrag(valp, ks*16 + nt, lane), c);
      int n = nt*16 + colb;
#pragma unroll
      for (int r = 0; r < 8; ++r)
        v_ws[(size_t)(e0 + r + hl*8) * 256 + n] = f2bf(c[r]);
    }
  }
}

// ---------------- softmax-weighted scatter (numerator + partition function) ---------------
__global__ void scatter_kernel(const float* __restrict__ la_ws, const uint16_t* __restrict__ v_ws,
                               const uint32_t* __restrict__ m_mono, const int* __restrict__ edge_dst,
                               float* __restrict__ z, float* __restrict__ attn){
  int e = blockIdx.x;
  int d = threadIdx.x;          // 0..255
  int h = d >> 5;
  int dst = edge_dst[e];
  float mval = mono2f(m_mono[(size_t)dst * 8 + h]);
  float p = __expf(la_ws[(size_t)e * 8 + h] - mval);
  if ((d & 31) == 0) unsafeAtomicAdd(&z[(size_t)dst * 8 + h], p);
  float vv = bf2f(v_ws[(size_t)e * 256 + d]);
  unsafeAtomicAdd(&attn[(size_t)dst * 256 + d], p * vv);
}

// ---------------- final projection per 16-node tile ---------------------------------------
__global__ void proj_kernel(const float* __restrict__ attn, const float* __restrict__ z,
                            const uint32_t* __restrict__ projp, float* __restrict__ out, int n_tiles){
  extern __shared__ char smem[];
  int lane = threadIdx.x & 31;
  int wv   = threadIdx.x >> 5;
  int tile = blockIdx.x * 2 + wv;
  if (tile >= n_tiles) return;
  int n0 = tile * 16;
  uint16_t* ab = (uint16_t*)(smem + (size_t)wv * 8192);
  for (int i = lane; i < 16*256; i += 32){
    int mr = i >> 8, c = i & 255, h = c >> 5;
    float zv = z[(size_t)(n0 + mr) * 8 + h];
    ab[i] = f2bf(attn[(size_t)(n0 + mr) * 256 + c] / (zv + 1e-30f));
  }
  lds_fence();
  bf16x16 af[8];
#pragma unroll
  for (int ks = 0; ks < 8; ++ks) af[ks] = gather_afrag((const uint32_t*)ab, 128, ks*32, lane);
  int colb = lane & 15, hl = lane >> 4;
#pragma unroll
  for (int nt = 0; nt < 16; ++nt){
    f32x8 c = {};
#pragma unroll
    for (int ks = 0; ks < 8; ++ks) c = wmma_bf16(af[ks], load_bfrag(projp, ks*16 + nt, lane), c);
    int n = nt*16 + colb;
#pragma unroll
    for (int r = 0; r < 8; ++r)
      out[(size_t)(n0 + r + hl*8) * 256 + n] = c[r];
  }
}

extern "C" void kernel_launch(void* const* d_in, const int* in_sizes, int n_in,
                              void* d_out, int out_size, void* d_ws, size_t ws_size,
                              hipStream_t stream){
  (void)n_in; (void)ws_size;
  const float* message      = (const float*)d_in[0];
  const float* edge_attr    = (const float*)d_in[1];
  const float* edge_scalars = (const float*)d_in[2];
  const float* w1    = (const float*)d_in[3];
  const float* b1    = (const float*)d_in[4];
  const float* w2    = (const float*)d_in[5];
  const float* b2    = (const float*)d_in[6];
  const float* w3    = (const float*)d_in[7];
  const float* W_ea  = (const float*)d_in[8];
  const float* W_alpha = (const float*)d_in[9];
  const float* W_lin = (const float*)d_in[10];
  const float* W_ea2 = (const float*)d_in[11];
  const float* W_val = (const float*)d_in[12];
  const float* alpha_dot = (const float*)d_in[13];
  const float* W_proj = (const float*)d_in[14];
  const int*   edge_dst = (const int*)d_in[15];

  const int E = in_sizes[0] / 256;
  const int N = out_size / 256;

  char* ws = (char*)d_ws;
  size_t off = 0;
  auto alloc = [&](size_t bytes)->char*{
    char* p = ws + off;
    off = (off + bytes + 255) & ~(size_t)255;
    return p;
  };
  uint16_t* v_ws   = (uint16_t*)alloc((size_t)E * 256 * 2);
  float*    la_ws  = (float*)   alloc((size_t)E * 8 * 4);
  uint32_t* m_mono = (uint32_t*)alloc((size_t)N * 8 * 4);
  float*    zbuf   = (float*)   alloc((size_t)N * 8 * 4);
  float*    attn   = (float*)   alloc((size_t)N * 256 * 4);
  uint32_t* w1p    = (uint32_t*)alloc(2048 * 4);
  uint32_t* w2p    = (uint32_t*)alloc(2048 * 4);
  uint32_t* w3p    = (uint32_t*)alloc(8192 * 4);
  uint32_t* eap    = (uint32_t*)alloc(4096 * 4);
  uint32_t* ea2p   = (uint32_t*)alloc(4096 * 4);
  uint32_t* alphap = (uint32_t*)alloc(8192 * 4);
  uint32_t* linp   = (uint32_t*)alloc(32768 * 4);
  uint32_t* valp   = (uint32_t*)alloc(32768 * 4);
  uint32_t* projp  = (uint32_t*)alloc(32768 * 4);

  auto pack = [&](const float* W, uint32_t* dst, int Kpad, int Nn, int Kreal){
    int total = (Kpad >> 5) * (Nn >> 4) * 256;
    pack_weight_kernel<<<(total + 255) / 256, 256, 0, stream>>>(W, dst, Kpad, Nn, Kreal);
  };
  pack(w1,      w1p,    64,  64,  64);
  pack(w2,      w2p,    64,  64,  64);
  pack(w3,      w3p,    64,  256, 64);
  pack(W_ea,    eap,    32,  256, 16);
  pack(W_ea2,   ea2p,   32,  256, 16);
  pack(W_alpha, alphap, 256, 64,  256);
  pack(W_lin,   linp,   256, 256, 256);
  pack(W_val,   valp,   256, 256, 256);
  pack(W_proj,  projp,  256, 256, 256);

  // zero m_mono, z, attn (contiguous span; must happen every call for graph replay)
  size_t zdw = (size_t)(((char*)attn + (size_t)N * 256 * 4) - (char*)m_mono) / 4;
  zero_kernel<<<(int)((zdw + 255) / 256), 256, 0, stream>>>(m_mono, (int)zdw);

  int n_tiles = E / 16;
  edge_kernel<<<(n_tiles + 1) / 2, 64, 2 * WV_BYTES, stream>>>(
      message, edge_attr, edge_scalars, b1, b2, alpha_dot, edge_dst,
      w1p, w2p, w3p, eap, ea2p, alphap, linp, valp,
      v_ws, la_ws, m_mono, n_tiles);

  scatter_kernel<<<E, 256, 0, stream>>>(la_ws, v_ws, m_mono, edge_dst, zbuf, attn);

  int ntilesN = N / 16;
  proj_kernel<<<(ntilesN + 1) / 2, 64, 2 * 8192, stream>>>(attn, zbuf, projp, (float*)d_out, ntilesN);
}
